// VectorQuantization_678604833366
// MI455X (gfx1250) — matrix operand order repaired and checked by reading, not compiled, not run on hardware
//
#include <hip/hip_runtime.h>
#include <hip/hip_bf16.h>

// ---- problem constants (from reference) ----
constexpr int N_PTS  = 262144;
constexpr int DIM    = 64;
constexpr int K_CODE = 1024;
constexpr int TILES  = K_CODE / 16;   // 64 code tiles of 16

typedef __attribute__((ext_vector_type(16))) __bf16 v16bf;
typedef __attribute__((ext_vector_type(8)))  float  v8f;

// ---------------------------------------------------------------------------
// Pre-pass 1: build WMMA B-matrix fragments for the codebook.
// B (32x16 per 16x16x32 step) k order used here (same as A): lane = N + 16*(Kfrag/16),
// VGPR v holds Kfrag = 16*(lane/16) + 2v, 2v+1.  One v16bf (32 B) per lane
// per (tile, chunk).  hi = rne(bf16), lo = rne(bf16(f - hi)).
// fragHi/fragLo: [tile][chunk][lane] of v16bf  -> 64*2*32*32 B = 128 KB each.
// ---------------------------------------------------------------------------
__global__ void vq_make_frags(const float* __restrict__ emb,
                              v16bf* __restrict__ fragHi,
                              v16bf* __restrict__ fragLo) {
    const int lane = threadIdx.x & 31;
    const int wv   = threadIdx.x >> 5;
    const int tile = blockIdx.x * 8 + wv;       // <<<8,256>>> => 64 waves
    const int n    = lane & 15;
    const int g    = lane >> 4;
#pragma unroll
    for (int c = 0; c < 2; ++c) {
        v16bf h, l;
#pragma unroll
        for (int v = 0; v < 8; ++v) {
            const int kb = 32 * c + 16 * g + 2 * v;           // D column
            const float2 ev = *(const float2*)(emb + (size_t)(tile * 16 + n) * DIM + kb);
            const __bf16 h0 = (__bf16)ev.x, h1 = (__bf16)ev.y;
            h[2 * v]     = h0;
            h[2 * v + 1] = h1;
            l[2 * v]     = (__bf16)(ev.x - (float)h0);
            l[2 * v + 1] = (__bf16)(ev.y - (float)h1);
        }
        fragHi[(tile * 2 + c) * 32 + lane] = h;
        fragLo[(tile * 2 + c) * 32 + lane] = l;
    }
}

// ---- Pre-pass 2: e2[k] = ||emb_k||^2 (exact fp32) ----
__global__ void vq_e2(const float* __restrict__ emb, float* __restrict__ e2) {
    const int k = blockIdx.x * blockDim.x + threadIdx.x;
    if (k < K_CODE) {
        float s = 0.f;
#pragma unroll 8
        for (int i = 0; i < DIM; ++i) { const float v = emb[k * DIM + i]; s += v * v; }
        e2[k] = s;
    }
}

// ---------------------------------------------------------------------------
// Main kernel: one wave per 16-row strip of x.  8 waves / block.
// A (16x32, 16-bit) k order used here (same as B): lanes 0-15 & 16-31 both hold M = lane%16;
// VGPR v holds K = 2v + 16*(lane/16), pairs.
// C layout: lane -> N = lane%16, VGPR v -> M = v + 8*(lane/16).
// ---------------------------------------------------------------------------
__global__ void __launch_bounds__(256)
vq_main(const float* __restrict__ x, const float* __restrict__ emb,
        const v16bf* __restrict__ fragHi, const v16bf* __restrict__ fragLo,
        const float* __restrict__ e2,
        float* __restrict__ outQ, float* __restrict__ outIdx) {
    __shared__ int sIdx[8 * 16];

    const int lane    = threadIdx.x & 31;
    const int wv      = threadIdx.x >> 5;
    const int rowBase = (blockIdx.x * 8 + wv) * 16;
    const int m       = lane & 15;
    const int g       = lane >> 4;

    // ---- build A fragments (hi/lo bf16 split) + per-lane ||x||^2 partial ----
    v16bf a_hi[2], a_lo[2];
    float x2p = 0.f;
#pragma unroll
    for (int c = 0; c < 2; ++c) {
#pragma unroll
        for (int v = 0; v < 8; ++v) {
            const int k0 = 2 * v + 16 * g;
            const float2 xv = *(const float2*)(x + (size_t)(rowBase + m) * DIM + 32 * c + k0);
            const __bf16 h0 = (__bf16)xv.x, h1 = (__bf16)xv.y;
            a_hi[c][2 * v]     = h0;
            a_hi[c][2 * v + 1] = h1;
            a_lo[c][2 * v]     = (__bf16)(xv.x - (float)h0);
            a_lo[c][2 * v + 1] = (__bf16)(xv.y - (float)h1);
            x2p += xv.x * xv.x + xv.y * xv.y;
        }
    }
    // lanes l and l^16 jointly cover all 64 dims of row m
    x2p += __shfl_xor(x2p, 16, 32);            // now: lane r holds ||x_{r%16}||^2

    float x2m[8];
#pragma unroll
    for (int v = 0; v < 8; ++v)
        x2m[v] = __shfl(x2p, v + 8 * g, 32);   // x2 for C-row M = v + 8g

    float best[8];
    int   bidx[8];
#pragma unroll
    for (int v = 0; v < 8; ++v) { best[v] = 3.4e38f; bidx[v] = 0; }

    // ---- sweep all 64 code tiles ----
    for (int t = 0; t < TILES; ++t) {
        if (t + 1 < TILES) {                   // warm L2/L0 for next tile's B frags
            __builtin_prefetch(&fragHi[((t + 1) * 2) * 32 + lane], 0, 3);
            __builtin_prefetch(&fragLo[((t + 1) * 2) * 32 + lane], 0, 3);
        }
        const float e2v = e2[t * 16 + m];
        v8f acc = {};
#pragma unroll
        for (int c = 0; c < 2; ++c) {
            const v16bf bh = fragHi[(t * 2 + c) * 32 + lane];
            const v16bf bl = fragLo[(t * 2 + c) * 32 + lane];
            acc = __builtin_amdgcn_wmma_f32_16x16x32_bf16(false, a_hi[c], false, bh,
                                                          (short)0, acc, false, false);
            acc = __builtin_amdgcn_wmma_f32_16x16x32_bf16(false, a_lo[c], false, bh,
                                                          (short)0, acc, false, false);
            acc = __builtin_amdgcn_wmma_f32_16x16x32_bf16(false, a_hi[c], false, bl,
                                                          (short)0, acc, false, false);
        }
#pragma unroll
        for (int v = 0; v < 8; ++v) {
            const float d2 = x2m[v] + e2v - 2.0f * acc[v];
            if (d2 < best[v]) { best[v] = d2; bidx[v] = t * 16 + m; }  // strict < keeps first min
        }
    }

    // ---- argmin across the 16 lanes of each half-wave (row M fixed per VGPR) ----
#pragma unroll
    for (int v = 0; v < 8; ++v) {
#pragma unroll
        for (int off = 8; off >= 1; off >>= 1) {
            const float ob = __shfl_xor(best[v], off, 32);
            const int   oi = __shfl_xor(bidx[v], off, 32);
            if (ob < best[v] || (ob == best[v] && oi < bidx[v])) { best[v] = ob; bidx[v] = oi; }
        }
    }
    if (m == 0) {                               // lanes 0 and 16 publish rows 0-7 / 8-15
#pragma unroll
        for (int v = 0; v < 8; ++v) sIdx[wv * 16 + v + 8 * g] = bidx[v];
    }
    __syncthreads();

    // ---- gather exact fp32 codebook rows + write indices ----
    for (int r = 0; r < 16; ++r) {
        const int idx = sIdx[wv * 16 + r];
        const float2 ev = *(const float2*)(emb + (size_t)idx * DIM + lane * 2);
        *(float2*)(outQ + (size_t)(rowBase + r) * DIM + lane * 2) = ev;
    }
    if (lane < 16) outIdx[rowBase + lane] = (float)sIdx[wv * 16 + lane];
}

// ---------------------------------------------------------------------------
extern "C" void kernel_launch(void* const* d_in, const int* in_sizes, int n_in,
                              void* d_out, int out_size, void* d_ws, size_t ws_size,
                              hipStream_t stream) {
    const float* x   = (const float*)d_in[0];   // [N, 64] fp32
    const float* emb = (const float*)d_in[1];   // [1024, 64] fp32

    float* outQ   = (float*)d_out;                       // quantized [N,64]
    float* outIdx = outQ + (size_t)N_PTS * DIM;          // indices [N] (as float)

    char*  ws     = (char*)d_ws;
    v16bf* fragHi = (v16bf*)(ws);                        // 128 KB
    v16bf* fragLo = (v16bf*)(ws + 131072);               // 128 KB
    float* e2     = (float*)(ws + 262144);               // 4 KB

    vq_make_frags<<<8, 256, 0, stream>>>(emb, fragHi, fragLo);
    vq_e2<<<4, 256, 0, stream>>>(emb, e2);
    vq_main<<<N_PTS / (16 * 8), 256, 0, stream>>>(x, emb, fragHi, fragLo, e2,
                                                  outQ, outIdx);
}
